// Allegro_Module_57930518888471
// MI455X (gfx1250) — compile-verified, matrix operand order checked by reading, not run on hardware
//
#include <hip/hip_runtime.h>
#include <hip/hip_bf16.h>
#include <math.h>

// ---------------- problem constants (match reference) ----------------
#define EDGES   160000
#define NODES   10000
#define NTYPES  4
#define NBASIS  8
#define UMULT   32          // env_embed_multiplicity
#define RMAX    6.0f
#define ENV_NORM 0.25f      // 1/sqrt(16)
#define INV_SQRT3 0.57735026918962576f

typedef __attribute__((ext_vector_type(16))) _Float16 v16h;
typedef __attribute__((ext_vector_type(8)))  _Float16 h8;
typedef __attribute__((ext_vector_type(8)))  float    v8f;

union Frag { v16h v; h8 h[2]; };

// =====================================================================
// WMMA GEMM:  out[M,N] = act( A[M,K] @ Bt^T )  with fused epilogue
//   A   : f16, row stride lda (>= K)
//   Bt  : f16, [N, K]  (transposed weight, pre-scaled by 1/sqrt(fan_in))
//   mode: 0 = plain, 1 = SiLU, 2 = *cut, 3 = *cut then resnet-combine
// One wave -> 32x64 tile (2 row-frags x 4 col-frags = 8 accumulators):
// 8 v_wmma per 12 b128 loads.  4 waves/block cover 32 rows x 256 cols.
// =====================================================================
__global__ __launch_bounds__(128)
void wmma_gemm(const _Float16* __restrict__ A, int lda,
               const _Float16* __restrict__ Bt, int K, int Nout,
               float* __restrict__ outF32, int ldF32,
               _Float16* __restrict__ outF16, int ldF16,
               const float* __restrict__ cutv,
               const float* __restrict__ resnetp, int resnetIdx,
               const float* __restrict__ oldF32, int ldOld,
               int mode)
{
    const int lane = threadIdx.x & 31;
    const int wave = threadIdx.x >> 5;
    const int l    = lane & 15;
    const int g    = lane >> 4;
    const int rowBase = blockIdx.x * 32;
    const int colBase = blockIdx.y * 256 + wave * 64;
    if (colBase >= Nout) return;                 // uniform per wave: EXEC stays all-ones

    // A fragment rows (16-bit 16x32 layout: lane group g holds
    // K = [kb+8g, kb+8g+8) and [kb+8g+16, kb+8g+24), contiguous 16B chunks)
    const _Float16* Ar0 = A + (size_t)(rowBase + l) * lda;
    const _Float16* Ar1 = Ar0 + (size_t)16 * lda;
    const _Float16* Br0 = Bt + (size_t)(colBase + l) * K;
    const _Float16* Br1 = Br0 + (size_t)16 * K;
    const _Float16* Br2 = Br0 + (size_t)32 * K;
    const _Float16* Br3 = Br0 + (size_t)48 * K;

    v8f acc[2][4];
    #pragma unroll
    for (int mi = 0; mi < 2; ++mi)
        #pragma unroll
        for (int ni = 0; ni < 4; ++ni)
            acc[mi][ni] = (v8f){};

    for (int kb = 0; kb < K; kb += 32) {
        const int k0 = kb + 8 * g;
        Frag a0, a1, b0, b1, b2, b3;
        a0.h[0] = *(const h8*)(Ar0 + k0);  a0.h[1] = *(const h8*)(Ar0 + k0 + 16);
        a1.h[0] = *(const h8*)(Ar1 + k0);  a1.h[1] = *(const h8*)(Ar1 + k0 + 16);
        b0.h[0] = *(const h8*)(Br0 + k0);  b0.h[1] = *(const h8*)(Br0 + k0 + 16);
        b1.h[0] = *(const h8*)(Br1 + k0);  b1.h[1] = *(const h8*)(Br1 + k0 + 16);
        b2.h[0] = *(const h8*)(Br2 + k0);  b2.h[1] = *(const h8*)(Br2 + k0 + 16);
        b3.h[0] = *(const h8*)(Br3 + k0);  b3.h[1] = *(const h8*)(Br3 + k0 + 16);
        acc[0][0] = __builtin_amdgcn_wmma_f32_16x16x32_f16(false, a0.v, false, b0.v, (short)0, acc[0][0], false, false);
        acc[0][1] = __builtin_amdgcn_wmma_f32_16x16x32_f16(false, a0.v, false, b1.v, (short)0, acc[0][1], false, false);
        acc[0][2] = __builtin_amdgcn_wmma_f32_16x16x32_f16(false, a0.v, false, b2.v, (short)0, acc[0][2], false, false);
        acc[0][3] = __builtin_amdgcn_wmma_f32_16x16x32_f16(false, a0.v, false, b3.v, (short)0, acc[0][3], false, false);
        acc[1][0] = __builtin_amdgcn_wmma_f32_16x16x32_f16(false, a1.v, false, b0.v, (short)0, acc[1][0], false, false);
        acc[1][1] = __builtin_amdgcn_wmma_f32_16x16x32_f16(false, a1.v, false, b1.v, (short)0, acc[1][1], false, false);
        acc[1][2] = __builtin_amdgcn_wmma_f32_16x16x32_f16(false, a1.v, false, b2.v, (short)0, acc[1][2], false, false);
        acc[1][3] = __builtin_amdgcn_wmma_f32_16x16x32_f16(false, a1.v, false, b3.v, (short)0, acc[1][3], false, false);
    }

    // ---------------- epilogue ----------------
    float cexp = 0.f, sOld = 0.f, rInv = 1.f;
    if (mode == 3) {
        cexp = __expf(resnetp[resnetIdx]);
        float n2 = 0.f;
        for (int j = 0; j < resnetIdx; ++j) { float cj = __expf(resnetp[j]); n2 += cj * cj; }
        sOld = sqrtf(n2);
        rInv = 1.0f / sqrtf(n2 + cexp * cexp);
    }
    // C/D layout: lanes 0-15 -> M 0..7 across VGPR 0..7, lanes 16-31 -> M 8..15
    #pragma unroll
    for (int mi = 0; mi < 2; ++mi) {
        const int row0 = rowBase + mi * 16 + g * 8;
        #pragma unroll
        for (int ni = 0; ni < 4; ++ni) {
            const int col = colBase + ni * 16 + l;
            #pragma unroll
            for (int r = 0; r < 8; ++r) {
                const int row = row0 + r;
                float x = acc[mi][ni][r];
                if (mode == 1) x = x / (1.f + __expf(-x));                   // SiLU
                if (mode >= 2) x *= cutv[row];                               // polynomial cutoff
                if (mode == 3) x = (sOld * oldF32[(size_t)row * ldOld + col] + cexp * x) * rInv;
                if (outF32) outF32[(size_t)row * ldF32 + col] = x;
                if (outF16) outF16[(size_t)row * ldF16 + col] = (_Float16)x;
            }
        }
    }
}

// ---------------- weight prep: Wt[n, k] = W[k, n] / sqrt(K), f16, zero-padded K ----
__global__ void prep_w(const float* __restrict__ W, _Float16* __restrict__ Wt,
                       int K, int N, int Kpad, float scale)
{
    int idx = blockIdx.x * blockDim.x + threadIdx.x;
    if (idx >= N * Kpad) return;
    int n = idx / Kpad, k = idx - n * Kpad;
    Wt[idx] = (k < K) ? (_Float16)(W[(size_t)k * N + n] * scale) : (_Float16)0.f;
}

// ---------------- edge input build + cutoff ----------------
__global__ void build_latin(const int* __restrict__ ei, const float* __restrict__ nattr,
                            const float* __restrict__ eemb, const float* __restrict__ elen,
                            _Float16* __restrict__ latin, float* __restrict__ cut)
{
    int e = blockIdx.x * blockDim.x + threadIdx.x;
    if (e >= EDGES) return;
    int c = ei[e], n = ei[EDGES + e];
    _Float16* o = latin + (size_t)e * 32;
    #pragma unroll
    for (int i = 0; i < 4; ++i) o[i]     = (_Float16)nattr[c * NTYPES + i];
    #pragma unroll
    for (int i = 0; i < 4; ++i) o[4 + i] = (_Float16)nattr[n * NTYPES + i];
    #pragma unroll
    for (int i = 0; i < 8; ++i) o[8 + i] = (_Float16)eemb[(size_t)e * NBASIS + i];
    #pragma unroll
    for (int i = 16; i < 32; ++i) o[i] = (_Float16)0.f;
    // poly cutoff p=6: 1 - 28 x^6 + 48 x^7 - 21 x^8
    float x  = elen[e] / RMAX;
    float x2 = x * x, x6 = x2 * x2 * x2;
    float f  = 1.f - 28.f * x6 + 48.f * x6 * x - 21.f * x6 * x2;
    cut[e] = (x < 1.f) ? f : 0.f;
}

__global__ void zero_f32(float* __restrict__ p, int n)
{
    int i = blockIdx.x * blockDim.x + threadIdx.x;
    if (i < n) p[i] = 0.f;
}

// ---------------- env embedding scatter: env[center] += edge_attr (x) w_env ----------
__global__ void scatter_env(const int* __restrict__ ei, const float* __restrict__ eattr,
                            const float* __restrict__ wsrc, int wstride, int woff,
                            float* __restrict__ env)
{
    int t = blockIdx.x * blockDim.x + threadIdx.x;
    if (t >= EDGES * UMULT) return;
    int e = t >> 5, u = t & 31;
    int c = ei[e];
    float w0 = wsrc[(size_t)e * wstride + woff + 2 * u];
    float w1 = wsrc[(size_t)e * wstride + woff + 2 * u + 1];
    const float* ea = eattr + (size_t)e * 4;
    float* dst = env + (size_t)c * (UMULT * 4) + u * 4;
    atomicAdd(dst + 0, ea[0] * w0);
    atomicAdd(dst + 1, ea[1] * w1);
    atomicAdd(dst + 2, ea[2] * w1);
    atomicAdd(dst + 3, ea[3] * w1);
}

// ---------------- layer-0 tensor product (per edge, per channel) ----------------
__global__ void tp0a(const int* __restrict__ ei, const float* __restrict__ eattr,
                     const float* __restrict__ w0, const float* __restrict__ env,
                     float* __restrict__ scal, float* __restrict__ xv,
                     _Float16* __restrict__ cat)
{
    int t = blockIdx.x * blockDim.x + threadIdx.x;
    if (t >= EDGES * UMULT) return;
    int e = t >> 5, u = t & 31;
    int c = ei[e];
    const float* ea = eattr + (size_t)e * 4;
    float wf0 = w0[(size_t)e * 128 + 2 * u];
    float wf1 = w0[(size_t)e * 128 + 2 * u + 1];
    float f0 = ea[0] * wf0, f1 = ea[1] * wf1, f2 = ea[2] * wf1, f3 = ea[3] * wf1;
    const float* ev = env + (size_t)c * 128 + u * 4;
    float e0 = ev[0] * ENV_NORM, e1 = ev[1] * ENV_NORM, e2 = ev[2] * ENV_NORM, e3 = ev[3] * ENV_NORM;
    float t0 = f0 * e0;
    float t1 = (f1 * e1 + f2 * e2 + f3 * e3) * INV_SQRT3;
    scal[(size_t)e * 64 + u]      = t0;
    scal[(size_t)e * 64 + 32 + u] = t1;
    cat[(size_t)e * 576 + 512 + u]       = (_Float16)t0;   // concat([latents, scalars]) tail
    cat[(size_t)e * 576 + 512 + 32 + u]  = (_Float16)t1;
    float* xve = xv + (size_t)e * 192;
    xve[u * 3 + 0] = f0 * e1;   xve[u * 3 + 1] = f0 * e2;   xve[u * 3 + 2] = f0 * e3;   // v0
    xve[(32 + u) * 3 + 0] = f1 * e0; xve[(32 + u) * 3 + 1] = f2 * e0; xve[(32 + u) * 3 + 2] = f3 * e0; // v1
}

// ---------------- e3nn Linear mixing: features = alpha * (scalars@Ws, xv@Wv) ----------
__global__ void tp0b(const float* __restrict__ scal, const float* __restrict__ xv,
                     const float* __restrict__ Ws, const float* __restrict__ Wv,
                     float* __restrict__ feat)
{
    int t = blockIdx.x * blockDim.x + threadIdx.x;
    if (t >= EDGES * UMULT) return;
    int e = t >> 5, o = t & 31;
    const float alpha = 0.125f;                 // 1/sqrt(2U)
    const float* s = scal + (size_t)e * 64;
    const float* x = xv   + (size_t)e * 192;
    float ys = 0.f, y0 = 0.f, y1 = 0.f, y2 = 0.f;
    for (int k = 0; k < 64; ++k) {
        float ws = Ws[k * 32 + o];
        float wv = Wv[k * 32 + o];
        ys += s[k] * ws;
        y0 += x[k * 3 + 0] * wv; y1 += x[k * 3 + 1] * wv; y2 += x[k * 3 + 2] * wv;
    }
    float* fo = feat + (size_t)e * 128 + o * 4;
    fo[0] = ys * alpha; fo[1] = y0 * alpha; fo[2] = y1 * alpha; fo[3] = y2 * alpha;
}

// ---------------- layer-1 tensor product: scalars only ----------------
__global__ void tp1(const int* __restrict__ ei, const float* __restrict__ feat,
                    const float* __restrict__ env, _Float16* __restrict__ cat)
{
    int t = blockIdx.x * blockDim.x + threadIdx.x;
    if (t >= EDGES * UMULT) return;
    int e = t >> 5, u = t & 31;
    int c = ei[e];
    const float* f  = feat + (size_t)e * 128 + u * 4;
    const float* ev = env  + (size_t)c * 128 + u * 4;
    float e0 = ev[0] * ENV_NORM, e1 = ev[1] * ENV_NORM, e2 = ev[2] * ENV_NORM, e3 = ev[3] * ENV_NORM;
    float t0 = f[0] * e0;
    float t1 = (f[1] * e1 + f[2] * e2 + f[3] * e3) * INV_SQRT3;
    cat[(size_t)e * 576 + 512 + u]      = (_Float16)t0;
    cat[(size_t)e * 576 + 512 + 32 + u] = (_Float16)t1;
}

// =====================================================================
extern "C" void kernel_launch(void* const* d_in, const int* in_sizes, int n_in,
                              void* d_out, int out_size, void* d_ws, size_t ws_size,
                              hipStream_t stream)
{
    (void)in_sizes; (void)n_in; (void)out_size; (void)ws_size;
    const int*   ei    = (const int*)  d_in[0];
    const float* nattr = (const float*)d_in[1];
    const float* eattr = (const float*)d_in[2];
    const float* eemb  = (const float*)d_in[3];
    const float* elen  = (const float*)d_in[4];
    const float* W2b0  = (const float*)d_in[5];
    const float* W2b1  = (const float*)d_in[6];
    const float* W2b2  = (const float*)d_in[7];
    const float* Wenv0 = (const float*)d_in[8];
    const float* Wlat0 = (const float*)d_in[9];
    const float* Wlat1 = (const float*)d_in[10];
    const float* Wenv1 = (const float*)d_in[11];
    const float* Wfin0 = (const float*)d_in[12];
    const float* Wfin1 = (const float*)d_in[13];
    const float* Wlin_s = (const float*)d_in[14];
    const float* Wlin_v = (const float*)d_in[15];
    const float* rp     = (const float*)d_in[16];
    float* out = (float*)d_out;

    // ---------- workspace carve (aliased where lifetimes allow) ----------
    char* ws = (char*)d_ws;
    size_t off = 0;
    auto carve = [&](size_t bytes) -> char* {
        char* p = ws + off;
        off = (off + bytes + 255) & ~(size_t)255;
        return p;
    };
    _Float16* latin = (_Float16*)carve((size_t)EDGES * 32 * 2);
    float*    cut   = (float*)   carve((size_t)EDGES * 4);
    _Float16* h1    = (_Float16*)carve((size_t)EDGES * 128 * 2);
    _Float16* h2    = (_Float16*)carve((size_t)EDGES * 256 * 2);  // aliased by h512
    float*    w0    = (float*)   carve((size_t)EDGES * 128 * 4);  // aliased by h512 tail
    _Float16* h512  = (_Float16*)h2;                              // [E,512] f16, reuses h2+w0
    float*    lat   = (float*)   carve((size_t)EDGES * 512 * 4);
    _Float16* cat   = (_Float16*)carve((size_t)EDGES * 576 * 2);
    float*    env   = (float*)   carve((size_t)NODES * 128 * 4);
    float*    scal  = (float*)   carve((size_t)EDGES * 64 * 4);   // aliased by wenv1o
    float*    wenv1o= (float*)   scal;
    float*    xv    = (float*)   carve((size_t)EDGES * 192 * 4);
    float*    feat  = (float*)   carve((size_t)EDGES * 128 * 4);
    // f16 transposed weights
    _Float16* W2b0t  = (_Float16*)carve((size_t)128 * 32  * 2);
    _Float16* W2b1t  = (_Float16*)carve((size_t)256 * 128 * 2);
    _Float16* W2b2t  = (_Float16*)carve((size_t)512 * 256 * 2);
    _Float16* Wenv0t = (_Float16*)carve((size_t)128 * 512 * 2);
    _Float16* Wlat0t = (_Float16*)carve((size_t)512 * 576 * 2);
    _Float16* Wlat1t = (_Float16*)carve((size_t)512 * 512 * 2);
    _Float16* Wenv1t = (_Float16*)carve((size_t)64  * 512 * 2);
    _Float16* Wfin0t = (_Float16*)carve((size_t)512 * 576 * 2);
    _Float16* Wfin1t = (_Float16*)carve((size_t)512 * 512 * 2);

    const int TB = 256;
    auto g1 = [&](int n) { return dim3((n + TB - 1) / TB); };

    // ---------- weight prep ----------
    prep_w<<<g1(128 * 32),  TB, 0, stream>>>(W2b0,  W2b0t,  16,  128, 32,  1.f / sqrtf(16.f));
    prep_w<<<g1(256 * 128), TB, 0, stream>>>(W2b1,  W2b1t,  128, 256, 128, 1.f / sqrtf(128.f));
    prep_w<<<g1(512 * 256), TB, 0, stream>>>(W2b2,  W2b2t,  256, 512, 256, 1.f / sqrtf(256.f));
    prep_w<<<g1(128 * 512), TB, 0, stream>>>(Wenv0, Wenv0t, 512, 128, 512, 1.f / sqrtf(512.f));
    prep_w<<<g1(512 * 576), TB, 0, stream>>>(Wlat0, Wlat0t, 576, 512, 576, 1.f / sqrtf(576.f));
    prep_w<<<g1(512 * 512), TB, 0, stream>>>(Wlat1, Wlat1t, 512, 512, 512, 1.f / sqrtf(512.f));
    prep_w<<<g1(64 * 512),  TB, 0, stream>>>(Wenv1, Wenv1t, 512, 64,  512, 1.f / sqrtf(512.f));
    prep_w<<<g1(512 * 576), TB, 0, stream>>>(Wfin0, Wfin0t, 576, 512, 576, 1.f / sqrtf(576.f));
    prep_w<<<g1(512 * 512), TB, 0, stream>>>(Wfin1, Wfin1t, 512, 512, 512, 1.f / sqrtf(512.f));

    // ---------- edge inputs + cutoff ----------
    build_latin<<<g1(EDGES), TB, 0, stream>>>(ei, nattr, eemb, elen, latin, cut);

    const dim3 GB(128);
    const int MT = EDGES / 32;          // 5000 row tiles (32 rows per block)
    auto gy = [](int Nout) { return dim3(0); (void)Nout; };
    (void)gy;

    // ---------- two-body MLP ----------
    wmma_gemm<<<dim3(MT, 1), GB, 0, stream>>>(latin, 32, W2b0t, 32, 128,
        nullptr, 0, h1, 128, nullptr, nullptr, -1, nullptr, 0, /*mode=*/1);
    wmma_gemm<<<dim3(MT, 1), GB, 0, stream>>>(h1, 128, W2b1t, 128, 256,
        nullptr, 0, h2, 256, nullptr, nullptr, -1, nullptr, 0, 1);
    wmma_gemm<<<dim3(MT, 2), GB, 0, stream>>>(h2, 256, W2b2t, 256, 512,
        lat, 512, cat, 576, cut, nullptr, -1, nullptr, 0, /*mode=2: *cut*/ 2);

    // ---------- env-embed layer 0 (feature + env weights) ----------
    wmma_gemm<<<dim3(MT, 1), GB, 0, stream>>>(cat, 576, Wenv0t, 512, 128,
        w0, 128, nullptr, 0, nullptr, nullptr, -1, nullptr, 0, 0);
    zero_f32<<<g1(NODES * 128), TB, 0, stream>>>(env, NODES * 128);
    scatter_env<<<g1(EDGES * UMULT), TB, 0, stream>>>(ei, eattr, w0, 128, 64, env);
    tp0a<<<g1(EDGES * UMULT), TB, 0, stream>>>(ei, eattr, w0, env, scal, xv, cat);
    tp0b<<<g1(EDGES * UMULT), TB, 0, stream>>>(scal, xv, Wlin_s, Wlin_v, feat);

    // ---------- latent resnet update ----------
    wmma_gemm<<<dim3(MT, 2), GB, 0, stream>>>(cat, 576, Wlat0t, 576, 512,
        nullptr, 0, h512, 512, nullptr, nullptr, -1, nullptr, 0, 1);
    wmma_gemm<<<dim3(MT, 2), GB, 0, stream>>>(h512, 512, Wlat1t, 512, 512,
        lat, 512, cat, 576, cut, rp, /*resnetIdx=*/1, lat, 512, /*mode=3*/ 3);

    // ---------- env-embed layer 1 + TP (scalars only) ----------
    wmma_gemm<<<dim3(MT, 1), GB, 0, stream>>>(cat, 576, Wenv1t, 512, 64,
        wenv1o, 64, nullptr, 0, nullptr, nullptr, -1, nullptr, 0, 0);
    zero_f32<<<g1(NODES * 128), TB, 0, stream>>>(env, NODES * 128);
    scatter_env<<<g1(EDGES * UMULT), TB, 0, stream>>>(ei, eattr, wenv1o, 64, 0, env);
    tp1<<<g1(EDGES * UMULT), TB, 0, stream>>>(ei, feat, env, cat);

    // ---------- final latent ----------
    wmma_gemm<<<dim3(MT, 2), GB, 0, stream>>>(cat, 576, Wfin0t, 576, 512,
        nullptr, 0, h512, 512, nullptr, nullptr, -1, nullptr, 0, 1);
    wmma_gemm<<<dim3(MT, 2), GB, 0, stream>>>(h512, 512, Wfin1t, 512, 512,
        out, 512, nullptr, 0, cut, rp, /*resnetIdx=*/2, lat, 512, 3);
}
